// MemoryRNN_34763465294054
// MI455X (gfx1250) — matrix operand order repaired and checked by reading, not compiled, not run on hardware
//
#include <hip/hip_runtime.h>
#include <hip/hip_bf16.h>
#include <stdint.h>

// Problem dimensions (fixed by the reference)
#define T_STEPS 256
#define B_SZ    32
#define DIN_SZ  4096
#define H_SZ    512
#define R_SZ    51
#define SIXH    (6 * H_SZ)   // 3072
#define FIVEH   (5 * H_SZ)   // 2560

// ---- vector types (raw clang ext vectors; safe inside unions) ----
typedef __attribute__((ext_vector_type(16))) __bf16    v16bf;
typedef __attribute__((ext_vector_type(8)))  float     v8f;
typedef __attribute__((ext_vector_type(4)))  float     f32x4;
typedef __attribute__((ext_vector_type(4)))  uint32_t  u32x4;
typedef __attribute__((ext_vector_type(2)))  uint32_t  u32x2;

union FragBF {
  uint32_t u[8];
  u32x4    q[2];
  v16bf    v;
};

__device__ __forceinline__ uint16_t f2bf(float f) {
  uint32_t u = __float_as_uint(f);
  u += 0x7FFFu + ((u >> 16) & 1u);          // round-to-nearest-even
  return (uint16_t)(u >> 16);
}
__device__ __forceinline__ uint32_t packbf2(float a, float b) {
  return (uint32_t)f2bf(a) | ((uint32_t)f2bf(b) << 16);
}
__device__ __forceinline__ float sigmf(float x) {
  return 1.0f / (1.0f + __expf(-x));
}

// --------------------------------------------------------------------------
// Kernel 0: convert W_state (2560x512) and W_out (51x512, zero-padded to 64
// rows) from f32 to bf16 into workspace.
// --------------------------------------------------------------------------
__global__ __launch_bounds__(256) void convert_weights(
    const float* __restrict__ W_state, const float* __restrict__ W_out,
    uint16_t* __restrict__ Wst, uint16_t* __restrict__ Wob) {
  int i = blockIdx.x * blockDim.x + threadIdx.x;
  const int nst = FIVEH * H_SZ;
  if (i < nst) {
    Wst[i] = f2bf(W_state[i]);
  } else {
    int j = i - nst;
    if (j < 64 * H_SZ) {
      int row = j / H_SZ;
      Wob[j] = f2bf(row < R_SZ ? W_out[j] : 0.0f);
    }
  }
}

// --------------------------------------------------------------------------
// Kernel 1: PI = X @ W_in^T + b_in   (8192x4096 @ 4096x3072 -> 8192x3072 f32)
// 256 threads (8 waves) per block, block tile 128x128, k-step 32.
// f32 -> bf16 conversion happens in-register during LDS staging.
// A/B fragments follow the CDNA5 wave32 16-bit WMMA layouts:
//   A (16x32): lanes<16 hold K{0..7,16..23}; lanes>=16 hold K{8..15,24..31}
//   B (32x16): lanes<16 hold K{0..7,16..23}; lanes>=16 hold K{8..15,24..31}
// LDS row stride = 24 dwords so the two u32x4 fragment loads are 16B aligned.
// --------------------------------------------------------------------------
__global__ __launch_bounds__(256) void pi_gemm(
    const float* __restrict__ X,      // 8192 x 4096
    const float* __restrict__ W_in,   // 3072 x 4096
    const float* __restrict__ b_in,   // 3072
    float* __restrict__ PI) {         // 8192 x 3072
  __shared__ uint32_t As[128 * 24];
  __shared__ uint32_t Bs[128 * 24];

  const int tid  = threadIdx.x;
  const int wave = tid >> 5;
  const int lane = tid & 31;
  const int half = lane >> 4;
  const int nl   = lane & 15;
  const int mBase = blockIdx.y * 128;
  const int nBase = blockIdx.x * 128;

  const v8f vzero = {0.f, 0.f, 0.f, 0.f, 0.f, 0.f, 0.f, 0.f};
  v8f acc[8];
#pragma unroll
  for (int j = 0; j < 8; ++j) acc[j] = vzero;

  float bi[8];
#pragma unroll
  for (int j = 0; j < 8; ++j) bi[j] = b_in[nBase + j * 16 + nl];

  for (int kk = 0; kk < DIN_SZ; kk += 32) {
    // Cooperative stage of A(128x32) and B(128x32) tiles, f32 -> packed bf16.
#pragma unroll
    for (int i = 0; i < 4; ++i) {
      int j   = tid + i * 256;   // 1024 float4 slots per tile
      int row = j >> 3;
      int c   = j & 7;
      f32x4 a = *(const f32x4*)(X + (size_t)(mBase + row) * DIN_SZ + kk + c * 4);
      u32x2 pa; pa.x = packbf2(a.x, a.y); pa.y = packbf2(a.z, a.w);
      *(u32x2*)&As[row * 24 + c * 2] = pa;
      f32x4 b = *(const f32x4*)(W_in + (size_t)(nBase + row) * DIN_SZ + kk + c * 4);
      u32x2 pb; pb.x = packbf2(b.x, b.y); pb.y = packbf2(b.z, b.w);
      *(u32x2*)&Bs[row * 24 + c * 2] = pb;
    }
    __syncthreads();

    FragBF af;
    {
      const uint32_t* p = &As[(wave * 16 + nl) * 24 + half * 4];
      af.q[0] = *(const u32x4*)(p);
      af.q[1] = *(const u32x4*)(p + 8);
    }
#pragma unroll
    for (int j = 0; j < 8; ++j) {
      FragBF bfr;
      const uint32_t* p = &Bs[(j * 16 + nl) * 24 + half * 4];
      bfr.q[0] = *(const u32x4*)(p);
      bfr.q[1] = *(const u32x4*)(p + 8);
      acc[j] = __builtin_amdgcn_wmma_f32_16x16x32_bf16(
          false, af.v, false, bfr.v, (short)0, acc[j], false, false);
    }
    __syncthreads();
  }

  // Write C tile (+ bias). C layout: VGPR r -> M=r (lanes<16) / M=r+8.
#pragma unroll
  for (int j = 0; j < 8; ++j) {
#pragma unroll
    for (int r = 0; r < 8; ++r) {
      int row = mBase + wave * 16 + half * 8 + r;
      int col = nBase + j * 16 + nl;
      PI[(size_t)row * SIXH + col] = acc[j][r] + bi[j];
    }
  }
}

// --------------------------------------------------------------------------
// Kernel 2: persistent single-workgroup recurrent scan.
// LDS: mem_c (51x512 f32, 104448B) + mem_h (51 rows x 520 bf16, 53040B,
// padded so fragment u32x4 loads stay 16B aligned) + 32 relation ids.
// Each of 32 waves owns a 16-column slice of H and computes the 5 gate
// pre-activations for both 16-batch row tiles with WMMA; A fragments are
// gathered straight through rel_ids (indirect LDS row base). 8 waves then
// compute the 32x51 prediction GEMM for the step.
// --------------------------------------------------------------------------
__global__ __launch_bounds__(1024) void scan_kernel(
    const int*   __restrict__ rel_ids,  // 256 x 32
    const float* __restrict__ mem_h0,   // 51 x 512
    const float* __restrict__ mem_c0,   // 51 x 512
    const float* __restrict__ b_state,  // 2560
    const float* __restrict__ b_out,    // 51
    const float* __restrict__ dropout,  // 32 x 512
    const float* __restrict__ PI,       // 8192 x 3072
    const uint16_t* __restrict__ Wst,   // 2560 x 512 bf16
    const uint16_t* __restrict__ Wob,   // 64 x 512 bf16 (rows >= 51 are 0)
    float* __restrict__ out) {          // 8192 x 51
  extern __shared__ __align__(16) uint8_t smem[];
  float*     memC  = (float*)smem;                      // 104448 B
  uint32_t*  memHd = (uint32_t*)(smem + 104448);        // 51 * 260 dwords
  uint16_t*  memHs = (uint16_t*)memHd;                  // stride 520 ushorts
  int*       idsBuf = (int*)(smem + 104448 + 53040);    // 32 ints

  const int tid  = threadIdx.x;
  const int wave = tid >> 5;
  const int lane = tid & 31;
  const int half = lane >> 4;
  const int nl   = lane & 15;
  const int col  = wave * 16 + nl;   // this wave's H column for lane

  // ---- init LDS state from mem_h0 / mem_c0 ----
  for (int i = tid; i < R_SZ * H_SZ; i += 1024) {
    memC[i] = mem_c0[i];
    int row = i >> 9, c = i & 511;
    memHs[row * 520 + c] = f2bf(mem_h0[i]);
  }
  __syncthreads();

  float bst[5];
#pragma unroll
  for (int g = 0; g < 5; ++g) bst[g] = b_state[g * H_SZ + col];

  const v8f vzero = {0.f, 0.f, 0.f, 0.f, 0.f, 0.f, 0.f, 0.f};

  for (int t = 0; t < T_STEPS; ++t) {
    if (tid < 32) idsBuf[tid] = rel_ids[t * 32 + tid];
    __syncthreads();

    float cnv[2][8], hnv[2][8];

    // ---- phase 1+2: per row-tile, ps GEMM then gate math (regs only) ----
#pragma unroll
    for (int tile = 0; tile < 2; ++tile) {
      const int idA = idsBuf[tile * 16 + nl];   // relation row for batch m=nl
      v8f acc[5];
#pragma unroll
      for (int g = 0; g < 5; ++g) acc[g] = vzero;

      for (int kt = 0; kt < 16; ++kt) {
        FragBF a;
        const uint32_t* pa = memHd + idA * 260 + kt * 16 + half * 4;
        a.q[0] = *(const u32x4*)pa;
        a.q[1] = *(const u32x4*)(pa + 8);
#pragma unroll
        for (int g = 0; g < 5; ++g) {
          FragBF bfr;
          const uint32_t* pb =
              (const uint32_t*)(Wst + (size_t)(g * H_SZ + col) * H_SZ) +
              kt * 16 + half * 4;
          bfr.q[0] = *(const u32x4*)pb;
          bfr.q[1] = *(const u32x4*)(pb + 8);
          acc[g] = __builtin_amdgcn_wmma_f32_16x16x32_bf16(
              false, a.v, false, bfr.v, (short)0, acc[g], false, false);
        }
      }

      // gate math for this tile's 8 batches per lane (f32 throughout)
#pragma unroll
      for (int r = 0; r < 8; ++r) {
        const int b  = tile * 16 + half * 8 + r;
        const int id = idsBuf[b];
        const float* prow = PI + (size_t)(t * 32 + b) * SIXH;
        float pii  = prow[col];
        float pif  = prow[H_SZ + col];
        float pig  = prow[2 * H_SZ + col];
        float pio  = prow[3 * H_SZ + col];
        float pihw = prow[4 * H_SZ + col];
        float pipr = prow[5 * H_SZ + col];
        float ig = sigmf(pii + acc[0][r] + bst[0]);
        float fg = sigmf(pif + acc[1][r] + bst[1]);
        float gg = tanhf(pig + acc[2][r] + bst[2]);
        float og = sigmf(pio + acc[3][r] + bst[3]);
        float hw = sigmf(pihw + acc[4][r] + bst[4]);
        float cprev = memC[id * H_SZ + col];
        float cn = ig * gg + fg * cprev;
        float hn = og * tanhf(cn);
        hn = hw * hn + (1.0f - hw) * pipr;
        hn *= dropout[b * H_SZ + col];
        cnv[tile][r] = cn;
        hnv[tile][r] = hn;
      }
    }
    __syncthreads();   // all mem_h fragment reads of this step are done

    // ---- phase 3: scatter new state back into LDS memory banks ----
#pragma unroll
    for (int tile = 0; tile < 2; ++tile) {
#pragma unroll
      for (int r = 0; r < 8; ++r) {
        const int b  = tile * 16 + half * 8 + r;
        const int id = idsBuf[b];
        memC[id * H_SZ + col]  = cnv[tile][r];
        memHs[id * 520 + col]  = f2bf(hnv[tile][r]);
      }
    }
    __syncthreads();

    // ---- phase 4: pred = h_new @ W_out^T + b_out  (waves 0..7) ----
    if (wave < 8) {
      const int mtile = wave & 1;       // batch tile
      const int ntile = wave >> 1;      // output-column tile (64 padded)
      const int idP   = idsBuf[mtile * 16 + nl];
      v8f accP = vzero;
      for (int kt = 0; kt < 16; ++kt) {
        FragBF a, bfr;
        const uint32_t* pa = memHd + idP * 260 + kt * 16 + half * 4;
        a.q[0] = *(const u32x4*)pa;
        a.q[1] = *(const u32x4*)(pa + 8);
        const uint32_t* pb =
            (const uint32_t*)(Wob + (size_t)(ntile * 16 + nl) * H_SZ) +
            kt * 16 + half * 4;
        bfr.q[0] = *(const u32x4*)pb;
        bfr.q[1] = *(const u32x4*)(pb + 8);
        accP = __builtin_amdgcn_wmma_f32_16x16x32_bf16(
            false, a.v, false, bfr.v, (short)0, accP, false, false);
      }
      const int colp = ntile * 16 + nl;
      if (colp < R_SZ) {
        const float bo = b_out[colp];
#pragma unroll
        for (int r = 0; r < 8; ++r) {
          int b = mtile * 16 + half * 8 + r;
          out[(size_t)(t * 32 + b) * R_SZ + colp] = accP[r] + bo;
        }
      }
    }
    __syncthreads();
  }
}

// --------------------------------------------------------------------------
extern "C" void kernel_launch(void* const* d_in, const int* in_sizes, int n_in,
                              void* d_out, int out_size, void* d_ws, size_t ws_size,
                              hipStream_t stream) {
  const float* x       = (const float*)d_in[0];
  const int*   rel_ids = (const int*)  d_in[1];
  const float* mem_h0  = (const float*)d_in[2];
  const float* mem_c0  = (const float*)d_in[3];
  const float* W_in    = (const float*)d_in[4];
  const float* b_in    = (const float*)d_in[5];
  const float* W_state = (const float*)d_in[6];
  const float* b_state = (const float*)d_in[7];
  const float* W_out   = (const float*)d_in[8];
  const float* b_out   = (const float*)d_in[9];
  const float* dropout = (const float*)d_in[10];

  // workspace layout: PI f32 (96MB) | W_state bf16 (2.5MB) | W_out bf16 (64KB)
  float*    PI  = (float*)d_ws;
  uint16_t* Wst = (uint16_t*)((char*)d_ws + (size_t)8192 * 3072 * 4);
  uint16_t* Wob = (uint16_t*)((char*)Wst + (size_t)FIVEH * H_SZ * 2);

  {
    int n = (FIVEH + 64) * H_SZ;
    convert_weights<<<(n + 255) / 256, 256, 0, stream>>>(W_state, W_out, Wst, Wob);
  }
  {
    dim3 grid(SIXH / 128, 8192 / 128);   // (24, 64)
    pi_gemm<<<grid, 256, 0, stream>>>(x, W_in, b_in, PI);
  }
  {
    size_t shmem = 104448 + 53040 + 128; // memC + memH(bf16, padded) + ids
    scan_kernel<<<1, 1024, shmem, stream>>>(rel_ids, mem_h0, mem_c0, b_state,
                                            b_out, dropout, PI, Wst, Wob,
                                            (float*)d_out);
  }
}
